// EfficientFormerV2AttentionFFN_14190571946626
// MI455X (gfx1250) — compile-verified, matrix operand order checked by reading, not run on hardware
//
#include <hip/hip_runtime.h>
#include <hip/hip_bf16.h>

typedef __attribute__((ext_vector_type(16))) _Float16 v16h;
typedef __attribute__((ext_vector_type(8)))  _Float16 v8h;
typedef __attribute__((ext_vector_type(8)))  float    v8f;

#define DEV __device__ __forceinline__

// ---------------- problem constants ----------------
#define B_   32
#define C_   384
#define R_   16
#define H_   8
#define KD_  32
#define DV_  128
#define HD_  1024      // H*DV
#define MH_  1536      // 4*C
#define N_   256       // R*R
#define HK_  256       // H*KD
#define QK_SCALE 0.17677669529663687f   // 1/sqrt(32)

// ---------------- workspace layout (bytes) ----------------
// all blocks 256B-aligned
static constexpr size_t OFF_XH    = 0;                      // f16 [B][N][C]        6,291,456
static constexpr size_t OFF_WQKV  = OFF_XH   + 6291456;     // f16 [1536][384]      1,179,648
static constexpr size_t OFF_BQKV  = OFF_WQKV + 1179648;     // f32 [1536]               6,144
static constexpr size_t OFF_QT    = OFF_BQKV + 6144;        // f16 [B][H][N][KD]    4,194,304
static constexpr size_t OFF_KT    = OFF_QT   + 4194304;     // f16 [B][H][N][KD]    4,194,304
static constexpr size_t OFF_VH    = OFF_KT   + 4194304;     // f16 [B][HD][N]      16,777,216
static constexpr size_t OFF_VLOC  = OFF_VH   + 16777216;    // f32 [B][HD][N]      33,554,432
static constexpr size_t OFF_BIASF = OFF_VLOC + 33554432;    // f32 [H][N][N]        2,097,152
static constexpr size_t OFF_ACT   = OFF_BIASF+ 2097152;     // f16 [B][N][HD]      16,777,216
static constexpr size_t OFF_PW    = OFF_ACT  + 16777216;    // f16 [C][HD]            786,432
static constexpr size_t OFF_PB    = OFF_PW   + 786432;      // f32 [C]                  2,048
static constexpr size_t OFF_HF    = OFF_PB   + 2048;        // f32 [B][C][N]       12,582,912
static constexpr size_t OFF_HH    = OFF_HF   + 12582912;    // f16 [B][N][C]        6,291,456
static constexpr size_t OFF_M1W   = OFF_HH   + 6291456;     // f16 [MH][C]          1,179,648
static constexpr size_t OFF_M1B   = OFF_M1W  + 1179648;     // f32 [MH]                 6,144
static constexpr size_t OFF_MH_   = OFF_M1B  + 6144;        // f16 [B][MH][N]      25,165,824
static constexpr size_t OFF_MDW   = OFF_MH_  + 25165824;    // f16 [B][N][MH]      25,165,824
static constexpr size_t OFF_M2W   = OFF_MDW  + 25165824;    // f16 [C][MH]          1,179,648
static constexpr size_t OFF_M2B   = OFF_M2W  + 1179648;     // f32 [C]                  2,048
// total ~157.4 MB

// ---------------- fragment helpers (CDNA5 WMMA layouts) ----------------
// A 16x32 f16 (MxK), row-major source, stride in halfs:
//   row = lane&15; lane-half hs: elements 0..7 -> K = 8*hs..8*hs+7,
//   elements 8..15 -> K = 16+8*hs .. 16+8*hs+7  (two contiguous 16B loads)
DEV v16h ldfrag(const _Float16* base, int stride) {
    int l = (int)(threadIdx.x & 31u);
    const _Float16* p = base + (size_t)(l & 15) * stride + ((l >> 4) << 3);
    v8h lo = *(const v8h*)(p);
    v8h hi = *(const v8h*)(p + 16);
    return __builtin_shufflevector(lo, hi, 0,1,2,3,4,5,6,7,8,9,10,11,12,13,14,15);
}

// Same pattern, but source is f32 (LDS attention scores) converted to f16.
DEV v16h ldfrag_f32(const float* base, int stride) {
    int l = (int)(threadIdx.x & 31u);
    const float* p = base + (size_t)(l & 15) * stride + ((l >> 4) << 3);
    v16h r;
#pragma unroll
    for (int e = 0; e < 8; ++e) {
        r[e]     = (_Float16)p[e];
        r[e + 8] = (_Float16)p[e + 16];
    }
    return r;
}

DEV v8f wmma16(v16h a, v16h b, v8f c) {
    return __builtin_amdgcn_wmma_f32_16x16x32_f16(false, a, false, b, (short)0, c, false, false);
}

DEV float gelu_exact(float x) { return 0.5f * x * (1.0f + erff(x * 0.70710678118654752f)); }

// one wave computes a 64(M) x 64(N) tile: per k-step 8 fragment loads feed 16 WMMAs
// A row-major [M][K] (rows m0..m0+63), B column-major [n][K] (cols n0..n0+63)
template<int K>
DEV void gemm_wave_64x64(const _Float16* __restrict__ A, const _Float16* __restrict__ Bc,
                         v8f acc[4][4]) {
    for (int kc = 0; kc < K; kc += 32) {
        if (kc + 32 < K) {
            __builtin_prefetch(A + kc + 32, 0, 1);
            __builtin_prefetch(Bc + kc + 32, 0, 1);
        }
        v16h af[4], bf[4];
#pragma unroll
        for (int mi = 0; mi < 4; ++mi) af[mi] = ldfrag(A + (size_t)(mi * 16) * K + kc, K);
#pragma unroll
        for (int t = 0; t < 4; ++t)    bf[t]  = ldfrag(Bc + (size_t)(t * 16) * K + kc, K);
#pragma unroll
        for (int mi = 0; mi < 4; ++mi) {
#pragma unroll
            for (int t = 0; t < 4; ++t) {
                acc[mi][t] = wmma16(af[mi], bf[t], acc[mi][t]);
            }
        }
    }
}

DEV void zero_acc(v8f acc[4][4]) {
#pragma unroll
    for (int mi = 0; mi < 4; ++mi)
#pragma unroll
        for (int t = 0; t < 4; ++t)
#pragma unroll
            for (int e = 0; e < 8; ++e) acc[mi][t][e] = 0.0f;
}

// ---------------- prep kernels ----------------
// x [B][C][N] f32  ->  Xh [B][N][C] f16 (column-major B operand)
__global__ void k_cvtx(const float* __restrict__ x, _Float16* __restrict__ Xh) {
    int idx = blockIdx.x * blockDim.x + threadIdx.x;
    if (idx >= B_ * C_ * N_) return;
    int n = idx & (N_ - 1);
    int c = (idx >> 8) % C_;
    int b = idx / (C_ * N_);
    Xh[((size_t)b * N_ + n) * C_ + c] = (_Float16)x[idx];
}

// fold BN affine into weights/bias: wh = s*w*extra ; bout = (s*b+sb)*extra
__global__ void k_fold(const float* __restrict__ w, const float* __restrict__ b,
                       const float* __restrict__ s, const float* __restrict__ sb,
                       _Float16* __restrict__ wh, float* __restrict__ bout,
                       int M, int K, float extra) {
    int idx = blockIdx.x * blockDim.x + threadIdx.x;
    if (idx < M * K) {
        int row = idx / K;
        wh[idx] = (_Float16)(s[row] * w[idx] * extra);
    }
    if (idx < M) bout[idx] = (s[idx] * b[idx] + sb[idx]) * extra;
}

// dense relative-position bias: bf[h][n][m] = attn_bias[h][bias_idxs[n][m]]
__global__ void k_biasfull(const float* __restrict__ ab, const int* __restrict__ idxs,
                           float* __restrict__ bf) {
    int idx = blockIdx.x * blockDim.x + threadIdx.x;
    if (idx >= H_ * N_ * N_) return;
    int h = idx >> 16;
    int r = idx & 65535;
    bf[idx] = ab[h * N_ + idxs[r]];
}

// ---------------- QKV GEMM ----------------
// grid(M/64, B); 4 waves per block, wave w owns token columns w*64..w*64+63
__global__ __launch_bounds__(128)
void k_gemm_qkv(const _Float16* __restrict__ Wh, const float* __restrict__ bias,
                const _Float16* __restrict__ Xh,
                _Float16* __restrict__ q_t, _Float16* __restrict__ k_t,
                _Float16* __restrict__ v_h) {
    int b  = blockIdx.y;
    int m0 = blockIdx.x * 64;
    int n0 = (int)(threadIdx.x >> 5) * 64;
    v8f acc[4][4];
    zero_acc(acc);
    gemm_wave_64x64<C_>(Wh + (size_t)m0 * C_, Xh + ((size_t)b * N_ + n0) * C_, acc);
    int l = (int)(threadIdx.x & 31u), hs = l >> 4, nl = l & 15;
#pragma unroll
    for (int mi = 0; mi < 4; ++mi) {
#pragma unroll
        for (int t = 0; t < 4; ++t) {
            int n = n0 + t * 16 + nl;
#pragma unroll
            for (int i = 0; i < 8; ++i) {
                int o = m0 + mi * 16 + i + 8 * hs;
                float v = acc[mi][t][i] + bias[o];
                if (o < HK_) {                       // Q -> [b][h][n][d]
                    int h = o >> 5, d = o & 31;
                    q_t[(((size_t)b * H_ + h) * N_ + n) * KD_ + d] = (_Float16)v;
                } else if (o < 2 * HK_) {            // K -> [b][h][m][d] (col-major B)
                    int o2 = o - HK_, h = o2 >> 5, d = o2 & 31;
                    k_t[(((size_t)b * H_ + h) * N_ + n) * KD_ + d] = (_Float16)v;
                } else {                             // V -> [b][o][n]  (natural & col-major B)
                    v_h[((size_t)b * HD_ + (o - 2 * HK_)) * N_ + n] = (_Float16)v;
                }
            }
        }
    }
}

// ---------------- depthwise 3x3 on V (+BN) ----------------
__global__ void k_dwv(const _Float16* __restrict__ v_h, const float* __restrict__ w,
                      const float* __restrict__ bb, const float* __restrict__ s,
                      const float* __restrict__ sb, float* __restrict__ v_local) {
    int idx = blockIdx.x * blockDim.x + threadIdx.x;
    if (idx >= B_ * HD_ * N_) return;
    int n = idx & (N_ - 1);
    int o = (idx >> 8) & (HD_ - 1);
    int b = idx >> 18;
    int r = n >> 4, c = n & 15;
    const _Float16* base = v_h + ((size_t)b * HD_ + o) * N_;
    const float* w9 = w + (size_t)o * 9;
    float a = bb[o];
#pragma unroll
    for (int dr = -1; dr <= 1; ++dr) {
        int rr = r + dr;
        if (rr < 0 || rr > 15) continue;
#pragma unroll
        for (int dc = -1; dc <= 1; ++dc) {
            int cc = c + dc;
            if (cc < 0 || cc > 15) continue;
            a += w9[(dr + 1) * 3 + (dc + 1)] * (float)base[rr * 16 + cc];
        }
    }
    v_local[idx] = a * s[o] + sb[o];
}

// ---------------- attention: QK^T + bias, talking heads, softmax, PV ----------------
// grid(16, 32): one WG per (16-row tile, batch). 8 waves = 8 heads. 128KB dyn LDS.
__global__ __launch_bounds__(256)
void k_attn(const _Float16* __restrict__ q_t, const _Float16* __restrict__ k_t,
            const _Float16* __restrict__ v_h, const float* __restrict__ v_local,
            const float* __restrict__ bias_full,
            const float* __restrict__ th1w, const float* __restrict__ th1b,
            const float* __restrict__ th2w, const float* __restrict__ th2b,
            _Float16* __restrict__ act_h) {
    extern __shared__ float SS_raw[];              // [8][16][256] f32
#define SS(h, i, m) SS_raw[(((h) << 4) + (i)) * 256 + (m)]
    int n0 = blockIdx.x * 16;
    int b  = blockIdx.y;
    int w  = (int)(threadIdx.x >> 5);              // wave id == head
    int l  = (int)(threadIdx.x & 31u);
    int hs = l >> 4, nl = l & 15;

    // ---- phase 1: S[h] = (q*scale) . k  + bias ----
    {
        v16h qa = ldfrag(q_t + (((size_t)b * H_ + w) * N_ + n0) * KD_, KD_);
        for (int mt = 0; mt < 16; ++mt) {
            v16h kb = ldfrag(k_t + (((size_t)b * H_ + w) * N_ + mt * 16) * KD_, KD_);
            v8f c = {0,0,0,0,0,0,0,0};
            c = wmma16(qa, kb, c);                 // KD=32: single WMMA step
            int m = mt * 16 + nl;
#pragma unroll
            for (int i = 0; i < 8; ++i) {
                int nr = i + 8 * hs;
                SS(w, nr, m) = c[i] + bias_full[((size_t)w * N_ + (n0 + nr)) * N_ + m];
            }
        }
    }
    __syncthreads();

    // ---- phase 2: talking head 1 (in place; each thread owns distinct (i,m)) ----
    for (int loc = (int)threadIdx.x; loc < 16 * N_; loc += 256) {
        int i = loc >> 8, m = loc & (N_ - 1);
        float sv[H_];
#pragma unroll
        for (int h = 0; h < H_; ++h) sv[h] = SS(h, i, m);
#pragma unroll
        for (int g = 0; g < H_; ++g) {
            float a = th1b[g];
#pragma unroll
            for (int h = 0; h < H_; ++h) a += th1w[g * H_ + h] * sv[h];
            SS(g, i, m) = a;
        }
    }
    __syncthreads();

    // ---- phase 3: softmax over m for each of 128 rows ----
    if (threadIdx.x < 128) {
        int g = (int)(threadIdx.x >> 4), i = (int)(threadIdx.x & 15u);
        float mx = -1e30f;
        for (int m = 0; m < N_; ++m) mx = fmaxf(mx, SS(g, i, m));
        float sum = 0.f;
        for (int m = 0; m < N_; ++m) { float e = expf(SS(g, i, m) - mx); SS(g, i, m) = e; sum += e; }
        float inv = 1.0f / sum;
        for (int m = 0; m < N_; ++m) SS(g, i, m) *= inv;
    }
    __syncthreads();

    // ---- phase 4: talking head 2 (in place) ----
    for (int loc = (int)threadIdx.x; loc < 16 * N_; loc += 256) {
        int i = loc >> 8, m = loc & (N_ - 1);
        float sv[H_];
#pragma unroll
        for (int h = 0; h < H_; ++h) sv[h] = SS(h, i, m);
#pragma unroll
        for (int g = 0; g < H_; ++g) {
            float a = th2b[g];
#pragma unroll
            for (int h = 0; h < H_; ++h) a += th2w[g * H_ + h] * sv[h];
            SS(g, i, m) = a;
        }
    }
    __syncthreads();

    // ---- phase 5: O = attn . V, + v_local, GELU, store act_h [b][n][o] f16 ----
    {
        v8f acc[8];
#pragma unroll
        for (int dt = 0; dt < 8; ++dt)
#pragma unroll
            for (int e = 0; e < 8; ++e) acc[dt][e] = 0.0f;
        for (int kc = 0; kc < N_; kc += 32) {
            v16h af = ldfrag_f32(&SS(w, 0, kc), N_);
#pragma unroll
            for (int dt = 0; dt < 8; ++dt) {
                v16h bf = ldfrag(v_h + ((size_t)b * HD_ + w * DV_ + dt * 16) * N_ + kc, N_);
                acc[dt] = wmma16(af, bf, acc[dt]);
            }
        }
#pragma unroll
        for (int dt = 0; dt < 8; ++dt) {
#pragma unroll
            for (int i = 0; i < 8; ++i) {
                int n = n0 + i + 8 * hs;
                int o = w * DV_ + dt * 16 + nl;
                float v = acc[dt][i] + v_local[((size_t)b * HD_ + o) * N_ + n];
                act_h[((size_t)b * N_ + n) * HD_ + o] = (_Float16)gelu_exact(v);
            }
        }
    }
#undef SS
}

// ---------------- projection GEMM + residual/layer-scale ----------------
__global__ __launch_bounds__(128)
void k_gemm_p(const _Float16* __restrict__ Wh, const float* __restrict__ bias,
              const _Float16* __restrict__ actB, const float* __restrict__ x,
              const float* __restrict__ ls1, float* __restrict__ h_f,
              _Float16* __restrict__ h_h) {
    int b  = blockIdx.y;
    int m0 = blockIdx.x * 64;
    int n0 = (int)(threadIdx.x >> 5) * 64;
    v8f acc[4][4];
    zero_acc(acc);
    gemm_wave_64x64<HD_>(Wh + (size_t)m0 * HD_, actB + ((size_t)b * N_ + n0) * HD_, acc);
    int l = (int)(threadIdx.x & 31u), hs = l >> 4, nl = l & 15;
#pragma unroll
    for (int mi = 0; mi < 4; ++mi) {
#pragma unroll
        for (int t = 0; t < 4; ++t) {
            int n = n0 + t * 16 + nl;
#pragma unroll
            for (int i = 0; i < 8; ++i) {
                int co = m0 + mi * 16 + i + 8 * hs;
                size_t nat = ((size_t)b * C_ + co) * N_ + n;
                float hv = x[nat] + ls1[co] * (acc[mi][t][i] + bias[co]);
                h_f[nat] = hv;
                h_h[((size_t)b * N_ + n) * C_ + co] = (_Float16)hv;
            }
        }
    }
}

// ---------------- MLP GEMM 1 (+GELU) ----------------
__global__ __launch_bounds__(128)
void k_gemm_m1(const _Float16* __restrict__ Wh, const float* __restrict__ bias,
               const _Float16* __restrict__ hB, _Float16* __restrict__ m_h) {
    int b  = blockIdx.y;
    int m0 = blockIdx.x * 64;
    int n0 = (int)(threadIdx.x >> 5) * 64;
    v8f acc[4][4];
    zero_acc(acc);
    gemm_wave_64x64<C_>(Wh + (size_t)m0 * C_, hB + ((size_t)b * N_ + n0) * C_, acc);
    int l = (int)(threadIdx.x & 31u), hs = l >> 4, nl = l & 15;
#pragma unroll
    for (int mi = 0; mi < 4; ++mi) {
#pragma unroll
        for (int t = 0; t < 4; ++t) {
            int n = n0 + t * 16 + nl;
#pragma unroll
            for (int i = 0; i < 8; ++i) {
                int mo = m0 + mi * 16 + i + 8 * hs;
                m_h[((size_t)b * MH_ + mo) * N_ + n] = (_Float16)gelu_exact(acc[mi][t][i] + bias[mo]);
            }
        }
    }
}

// ---------------- MLP depthwise 3x3 (+BN +GELU), transposed f16 store ----------------
__global__ void k_dwm(const _Float16* __restrict__ m_h, const float* __restrict__ w,
                      const float* __restrict__ bb, const float* __restrict__ s,
                      const float* __restrict__ sb, _Float16* __restrict__ mdw) {
    int idx = blockIdx.x * blockDim.x + threadIdx.x;
    if (idx >= B_ * MH_ * N_) return;
    int n = idx & (N_ - 1);
    int mo = (idx >> 8) % MH_;
    int b = idx / (MH_ * N_);
    int r = n >> 4, c = n & 15;
    const _Float16* base = m_h + ((size_t)b * MH_ + mo) * N_;
    const float* w9 = w + (size_t)mo * 9;
    float a = bb[mo];
#pragma unroll
    for (int dr = -1; dr <= 1; ++dr) {
        int rr = r + dr;
        if (rr < 0 || rr > 15) continue;
#pragma unroll
        for (int dc = -1; dc <= 1; ++dc) {
            int cc = c + dc;
            if (cc < 0 || cc > 15) continue;
            a += w9[(dr + 1) * 3 + (dc + 1)] * (float)base[rr * 16 + cc];
        }
    }
    a = a * s[mo] + sb[mo];
    mdw[((size_t)b * N_ + n) * MH_ + mo] = (_Float16)gelu_exact(a);
}

// ---------------- MLP GEMM 2 + final residual ----------------
__global__ __launch_bounds__(128)
void k_gemm_m2(const _Float16* __restrict__ Wh, const float* __restrict__ bias,
               const _Float16* __restrict__ mB, const float* __restrict__ h_f,
               const float* __restrict__ ls2, float* __restrict__ out) {
    int b  = blockIdx.y;
    int m0 = blockIdx.x * 64;
    int n0 = (int)(threadIdx.x >> 5) * 64;
    v8f acc[4][4];
    zero_acc(acc);
    gemm_wave_64x64<MH_>(Wh + (size_t)m0 * MH_, mB + ((size_t)b * N_ + n0) * MH_, acc);
    int l = (int)(threadIdx.x & 31u), hs = l >> 4, nl = l & 15;
#pragma unroll
    for (int mi = 0; mi < 4; ++mi) {
#pragma unroll
        for (int t = 0; t < 4; ++t) {
            int n = n0 + t * 16 + nl;
#pragma unroll
            for (int i = 0; i < 8; ++i) {
                int co = m0 + mi * 16 + i + 8 * hs;
                size_t nat = ((size_t)b * C_ + co) * N_ + n;
                out[nat] = h_f[nat] + ls2[co] * (acc[mi][t][i] + bias[co]);
            }
        }
    }
}

// ---------------- host launch ----------------
extern "C" void kernel_launch(void* const* d_in, const int* in_sizes, int n_in,
                              void* d_out, int out_size, void* d_ws, size_t ws_size,
                              hipStream_t stream) {
    (void)in_sizes; (void)n_in; (void)out_size; (void)ws_size;
    const float* x     = (const float*)d_in[0];
    const int*   bidx  = (const int*)  d_in[1];
    const float* q_w   = (const float*)d_in[2];
    const float* q_b   = (const float*)d_in[3];
    const float* q_s   = (const float*)d_in[4];
    const float* q_sb  = (const float*)d_in[5];
    const float* k_w   = (const float*)d_in[6];
    const float* k_b   = (const float*)d_in[7];
    const float* k_s   = (const float*)d_in[8];
    const float* k_sb  = (const float*)d_in[9];
    const float* v_w   = (const float*)d_in[10];
    const float* v_b   = (const float*)d_in[11];
    const float* v_s   = (const float*)d_in[12];
    const float* v_sb  = (const float*)d_in[13];
    const float* vl_w  = (const float*)d_in[14];
    const float* vl_b  = (const float*)d_in[15];
    const float* vl_s  = (const float*)d_in[16];
    const float* vl_sb = (const float*)d_in[17];
    const float* th1_w = (const float*)d_in[18];
    const float* th1_b = (const float*)d_in[19];
    const float* th2_w = (const float*)d_in[20];
    const float* th2_b = (const float*)d_in[21];
    const float* ab    = (const float*)d_in[22];
    const float* p_w   = (const float*)d_in[23];
    const float* p_b   = (const float*)d_in[24];
    const float* p_s   = (const float*)d_in[25];
    const float* p_sb  = (const float*)d_in[26];
    const float* ls1   = (const float*)d_in[27];
    const float* ls2   = (const float*)d_in[28];
    const float* m1_w  = (const float*)d_in[29];
    const float* m1_b  = (const float*)d_in[30];
    const float* m1_s  = (const float*)d_in[31];
    const float* m1_sb = (const float*)d_in[32];
    const float* mid_w = (const float*)d_in[33];
    const float* mid_b = (const float*)d_in[34];
    const float* mid_s = (const float*)d_in[35];
    const float* mid_sb= (const float*)d_in[36];
    const float* m2_w  = (const float*)d_in[37];
    const float* m2_b  = (const float*)d_in[38];
    const float* m2_s  = (const float*)d_in[39];
    const float* m2_sb = (const float*)d_in[40];
    float* out = (float*)d_out;

    char* ws = (char*)d_ws;
    _Float16* Xh     = (_Float16*)(ws + OFF_XH);
    _Float16* Wqkv   = (_Float16*)(ws + OFF_WQKV);
    float*    bqkv   = (float*)   (ws + OFF_BQKV);
    _Float16* q_t    = (_Float16*)(ws + OFF_QT);
    _Float16* k_t    = (_Float16*)(ws + OFF_KT);
    _Float16* v_h    = (_Float16*)(ws + OFF_VH);
    float*    v_loc  = (float*)   (ws + OFF_VLOC);
    float*    biasF  = (float*)   (ws + OFF_BIASF);
    _Float16* act_h  = (_Float16*)(ws + OFF_ACT);
    _Float16* p_wh   = (_Float16*)(ws + OFF_PW);
    float*    p_bf   = (float*)   (ws + OFF_PB);
    float*    h_f    = (float*)   (ws + OFF_HF);
    _Float16* h_h    = (_Float16*)(ws + OFF_HH);
    _Float16* m1_wh  = (_Float16*)(ws + OFF_M1W);
    float*    m1_bf  = (float*)   (ws + OFF_M1B);
    _Float16* m_h    = (_Float16*)(ws + OFF_MH_);
    _Float16* mdw    = (_Float16*)(ws + OFF_MDW);
    _Float16* m2_wh  = (_Float16*)(ws + OFF_M2W);
    float*    m2_bf  = (float*)   (ws + OFF_M2B);

    // prep
    k_cvtx<<<(B_*C_*N_ + 255)/256, 256, 0, stream>>>(x, Xh);
    k_fold<<<(HK_*C_ + 255)/256, 256, 0, stream>>>(q_w, q_b, q_s, q_sb, Wqkv,            bqkv,        HK_, C_,  QK_SCALE);
    k_fold<<<(HK_*C_ + 255)/256, 256, 0, stream>>>(k_w, k_b, k_s, k_sb, Wqkv + HK_*C_,   bqkv + HK_,  HK_, C_,  1.0f);
    k_fold<<<(HD_*C_ + 255)/256, 256, 0, stream>>>(v_w, v_b, v_s, v_sb, Wqkv + 2*HK_*C_, bqkv + 2*HK_, HD_, C_, 1.0f);
    k_fold<<<(C_*HD_ + 255)/256, 256, 0, stream>>>(p_w,  p_b,  p_s,  p_sb,  p_wh,  p_bf,  C_,  HD_, 1.0f);
    k_fold<<<(MH_*C_ + 255)/256, 256, 0, stream>>>(m1_w, m1_b, m1_s, m1_sb, m1_wh, m1_bf, MH_, C_,  1.0f);
    k_fold<<<(C_*MH_ + 255)/256, 256, 0, stream>>>(m2_w, m2_b, m2_s, m2_sb, m2_wh, m2_bf, C_,  MH_, 1.0f);
    k_biasfull<<<(H_*N_*N_ + 255)/256, 256, 0, stream>>>(ab, bidx, biasF);

    // token mixer
    k_gemm_qkv<<<dim3((2*HK_ + HD_)/64, B_), 128, 0, stream>>>(Wqkv, bqkv, Xh, q_t, k_t, v_h);
    k_dwv<<<(B_*HD_*N_ + 255)/256, 256, 0, stream>>>(v_h, vl_w, vl_b, vl_s, vl_sb, v_loc);
    k_attn<<<dim3(N_/16, B_), 256, 8*16*256*sizeof(float), stream>>>(
        q_t, k_t, v_h, v_loc, biasF, th1_w, th1_b, th2_w, th2_b, act_h);
    k_gemm_p<<<dim3(C_/64, B_), 128, 0, stream>>>(p_wh, p_bf, act_h, x, ls1, h_f, h_h);

    // ConvMLP
    k_gemm_m1<<<dim3(MH_/64, B_), 128, 0, stream>>>(m1_wh, m1_bf, h_h, m_h);
    k_dwm<<<(B_*MH_*N_ + 255)/256, 256, 0, stream>>>(m_h, mid_w, mid_b, mid_s, mid_sb, mdw);
    k_gemm_m2<<<dim3(C_/64, B_), 128, 0, stream>>>(m2_wh, m2_bf, mdw, h_f, ls2, out);
}
